// MultiheadAttention_51127290691872
// MI455X (gfx1250) — matrix operand order repaired and checked by reading, not compiled, not run on hardware
//
#include <hip/hip_runtime.h>
#include <hip/hip_bf16.h>

// ---------------------------------------------------------------------------
// MHA on gfx1250 (wave32): all matmuls via v_wmma_f32_16x16x32_f16.
// Staging uses gfx1250 async loads to LDS (global_load_async_to_lds_b128)
// with double-buffered tiles, tracked by ASYNCcnt.
// Pipeline: convert/transpose(f32->f16) -> QKV gemm -> flash attn -> out gemm.
// ---------------------------------------------------------------------------

typedef _Float16 f16;
typedef __attribute__((ext_vector_type(8)))  _Float16 v8h;
typedef __attribute__((ext_vector_type(16))) _Float16 v16h;
typedef __attribute__((ext_vector_type(8)))  float    v8f;

#define EMBED 1024
#define HEADS 16
#define DH    64
#define SEQ   2048
#define BATCH 2
#define MROWS (BATCH*SEQ)       // 4096

__device__ inline v16h cat8(v8h lo, v8h hi) {
    union { v16h v; v8h h[2]; } u;
    u.h[0] = lo; u.h[1] = hi;
    return u.v;
}

// D = A(16x32 f16) * B(32x16 f16) + C(f32)
__device__ inline v8f wmma32(v16h a, v16h b, v8f c) {
    return __builtin_amdgcn_wmma_f32_16x16x32_f16(
        /*neg_a=*/false, a, /*neg_b=*/false, b,
        /*c_mod=*/(short)0, c, /*reuse_a=*/false, /*reuse_b=*/false);
}

// A fragment: 16x32, row-major source, leading dim ld (elements).
// lane L: row = base_row + L%16; elems e: K = (e/8)*16 + (L/16)*8 + e%8.
__device__ inline v16h load_a_frag(const f16* base, int ld, int row, int k0,
                                   int hl) {
    const f16* p = base + (size_t)row * ld + k0 + hl * 8;
    v8h lo = *(const v8h*)p;
    v8h hi = *(const v8h*)(p + 16);
    return cat8(lo, hi);
}

// B fragment: 32x16, source stored TRANSPOSED as [N][K], leading dim ldk.
// lane L: n = base_n + L%16; elems e: K = (e/8)*16 + (L/16)*8 + e%8.
__device__ inline v16h load_b_frag(const f16* baseT, int ldk, int n, int k0,
                                   int hl) {
    const f16* p = baseT + (size_t)n * ldk + k0 + hl * 8;
    return cat8(*(const v8h*)p, *(const v8h*)(p + 16));
}

// ---- gfx1250 async copy: 16 bytes global -> LDS per lane (ASYNCcnt) -------
__device__ inline void async_copy_b128(const f16* gsrc, f16* lds_dst) {
    unsigned lds_off = (unsigned)(uintptr_t)lds_dst;   // LDS byte offset
    asm volatile("global_load_async_to_lds_b128 %0, %1, off"
                 :: "v"(lds_off), "v"(gsrc)
                 : "memory");
}

__device__ inline void async_wait0() {
#if __has_builtin(__builtin_amdgcn_s_wait_asynccnt)
    __builtin_amdgcn_s_wait_asynccnt(0);
#else
    asm volatile("s_wait_asynccnt 0x0" ::: "memory");
#endif
}

// ---------------------------------------------------------------------------
__global__ __launch_bounds__(256) void f32_to_f16_kernel(
    const float* __restrict__ src, f16* __restrict__ dst, int n) {
    int i = blockIdx.x * blockDim.x + threadIdx.x;
    if (i < n) dst[i] = (f16)src[i];
}

// W[K][N] f32 -> Wt[N][K] f16  (so GEMM B tiles are straight row-major copies)
__global__ __launch_bounds__(256) void transpose_f32_to_f16_kernel(
    const float* __restrict__ src, f16* __restrict__ dst, int Kdim, int N) {
    int i = blockIdx.x * blockDim.x + threadIdx.x;
    if (i < Kdim * N) {
        int k = i / N, n = i - k * N;
        dst[(size_t)n * Kdim + k] = (f16)src[i];
    }
}

// ---------------------------------------------------------------------------
// GEMM core: block tile 128x128, K-step 32, 8 waves (2x4), wave tile 64x32.
// A [M][K] row-major, B given pre-transposed Wt [N][K] row-major.
// Double-buffered LDS, async staging.
// ---------------------------------------------------------------------------
#define TM 128
#define TN 128
#define TK 32

// stages one 128x32 A tile and one 128x32 Bt tile into LDS buffer `buf`
__device__ inline void stage_tiles(const f16* X, const f16* Wt, int Kdim,
                                   int bm, int bn, int k0, int tid,
                                   f16 (*As)[TM][TK], f16 (*Bs)[TN][TK],
                                   int buf) {
#pragma unroll
    for (int c = tid; c < (TM * TK / 8); c += 256) {
        int r = c >> 2, kk = (c & 3) * 8;
        async_copy_b128(X + (size_t)(bm + r) * Kdim + k0 + kk,
                        &As[buf][r][kk]);
    }
#pragma unroll
    for (int c = tid; c < (TN * TK / 8); c += 256) {
        int n = c >> 2, kk = (c & 3) * 8;
        async_copy_b128(Wt + (size_t)(bn + n) * Kdim + k0 + kk,
                        &Bs[buf][n][kk]);
    }
}

// ---------------------------------------------------------------------------
// QKV projection: X[4096,1024] @ Wqkv^T-pretransposed + b -> Q/K/Vt f16.
// Q,K laid out [B,H,S,DH]; V stored TRANSPOSED as [B,H,DH,S].
// ---------------------------------------------------------------------------
__global__ __launch_bounds__(256) void gemm_qkv_kernel(
    const f16* __restrict__ X, const f16* __restrict__ Wt,
    const float* __restrict__ bias,
    f16* __restrict__ Qd, f16* __restrict__ Kd, f16* __restrict__ Vtd,
    int Kdim, int N) {
    __shared__ f16 As[2][TM][TK];
    __shared__ f16 Bs[2][TN][TK];

    const int tid = threadIdx.x;
    const int lane = tid & 31, wave = tid >> 5;
    const int hl = lane >> 4, l16 = lane & 15;
    const int wm = wave >> 2, wn = wave & 3;          // 2 x 4 waves
    const int bm = blockIdx.x * TM, bn = blockIdx.y * TN;

    v8f acc[4][2];
    const v8f vzero = {0.f, 0.f, 0.f, 0.f, 0.f, 0.f, 0.f, 0.f};
#pragma unroll
    for (int i = 0; i < 4; i++)
#pragma unroll
        for (int j = 0; j < 2; j++) acc[i][j] = vzero;

    stage_tiles(X, Wt, Kdim, bm, bn, 0, tid, As, Bs, 0);

    for (int k0 = 0; k0 < Kdim; k0 += TK) {
        const int cur = (k0 / TK) & 1;
        async_wait0();
        __syncthreads();
        if (k0 + TK < Kdim)
            stage_tiles(X, Wt, Kdim, bm, bn, k0 + TK, tid, As, Bs, cur ^ 1);

        v16h af[4], bf[2];
#pragma unroll
        for (int mt = 0; mt < 4; mt++)
            af[mt] = load_a_frag(&As[cur][0][0], TK,
                                 wm * 64 + mt * 16 + l16, 0, hl);
#pragma unroll
        for (int nt = 0; nt < 2; nt++)
            bf[nt] = load_b_frag(&Bs[cur][0][0], TK,
                                 wn * 32 + nt * 16 + l16, 0, hl);
#pragma unroll
        for (int mt = 0; mt < 4; mt++)
#pragma unroll
            for (int nt = 0; nt < 2; nt++)
                acc[mt][nt] = wmma32(af[mt], bf[nt], acc[mt][nt]);
    }

    // epilogue: col -> (head, q/k/v, d); row -> (b, s)
#pragma unroll
    for (int mt = 0; mt < 4; mt++)
#pragma unroll
        for (int nt = 0; nt < 2; nt++) {
            int col = bn + wn * 32 + nt * 16 + l16;
            float bv = bias[col];
            int h = col / (3 * DH);
            int j = col - h * (3 * DH);
            int which = j >> 6;      // 0=q 1=k 2=v
            int d = j & (DH - 1);
#pragma unroll
            for (int r = 0; r < 8; r++) {
                int row = bm + wm * 64 + mt * 16 + r + 8 * hl;
                int b = row >> 11;          // / SEQ
                int s = row & (SEQ - 1);
                float val = acc[mt][nt][r] + bv;
                size_t bh = (size_t)(b * HEADS + h);
                if (which == 0)
                    Qd[(bh * SEQ + s) * DH + d] = (f16)val;
                else if (which == 1)
                    Kd[(bh * SEQ + s) * DH + d] = (f16)val;
                else
                    Vtd[(bh * DH + d) * SEQ + s] = (f16)val;   // transposed
            }
        }
}

// ---------------------------------------------------------------------------
// Flash attention: grid (B*H, S/128). 8 waves x 16 q-rows. 64-key blocks,
// double-buffered async K / V^T tiles.
// ---------------------------------------------------------------------------
__device__ inline void stage_kv(const f16* Kb, const f16* Vtb, int kb, int tid,
                                f16 (*Ks)[64][DH], f16 (*Vts)[DH][64],
                                int buf) {
#pragma unroll
    for (int c = tid; c < 512; c += 256) {          // K tile [key][dh]
        int r = c >> 3, kk = (c & 7) * 8;
        async_copy_b128(Kb + (size_t)(kb + r) * DH + kk, &Ks[buf][r][kk]);
    }
#pragma unroll
    for (int c = tid; c < 512; c += 256) {          // V^T tile [dh][key]
        int d = c >> 3, kk = (c & 7) * 8;
        async_copy_b128(Vtb + (size_t)d * SEQ + kb + kk, &Vts[buf][d][kk]);
    }
}

__global__ __launch_bounds__(256) void attn_kernel(
    const f16* __restrict__ Q, const f16* __restrict__ K,
    const f16* __restrict__ Vt, f16* __restrict__ vals) {
    __shared__ f16 Ks[2][64][DH];     // [key][dh] == B-layout [N][K] for QK^T
    __shared__ f16 Vts[2][DH][64];    // [dh][key] == B-layout [N][K] for PV
    __shared__ f16 Ps[8][16][64];     // per-wave probs, A-layout source

    const int bh = blockIdx.x;
    const int qblk = blockIdx.y;
    const f16* Qb  = Q  + (size_t)bh * SEQ * DH;
    const f16* Kb  = K  + (size_t)bh * SEQ * DH;
    const f16* Vtb = Vt + (size_t)bh * DH * SEQ;

    const int tid = threadIdx.x;
    const int lane = tid & 31, wave = tid >> 5;
    const int hl = lane >> 4, l16 = lane & 15;

    // Q fragments for this wave's 16 rows (A-matrix: lane row = l16)
    const int qrowA = qblk * 128 + wave * 16 + l16;
    v16h qf[2];
#pragma unroll
    for (int kc = 0; kc < 2; kc++)
        qf[kc] = load_a_frag(Qb, DH, qrowA, kc * 32, hl);

    float m[8], l[8];
    v8f o[4];
    const v8f vzero = {0.f, 0.f, 0.f, 0.f, 0.f, 0.f, 0.f, 0.f};
#pragma unroll
    for (int r = 0; r < 8; r++) { m[r] = -__builtin_inff(); l[r] = 0.f; }
#pragma unroll
    for (int c = 0; c < 4; c++) o[c] = vzero;

    const float scale = 0.125f;   // 1/sqrt(64)

    stage_kv(Kb, Vtb, 0, tid, Ks, Vts, 0);

    for (int kb = 0; kb < SEQ; kb += 64) {
        const int cur = (kb >> 6) & 1;
        async_wait0();
        __syncthreads();
        if (kb + 64 < SEQ)
            stage_kv(Kb, Vtb, kb + 64, tid, Ks, Vts, cur ^ 1);

        // logits: 4 key-tiles of 16, accumulate over dh (2 chunks of 32)
        v8f sacc[4];
#pragma unroll
        for (int nt = 0; nt < 4; nt++) {
            sacc[nt] = vzero;
#pragma unroll
            for (int kc = 0; kc < 2; kc++) {
                v16h bf = load_b_frag(&Ks[cur][0][0], DH,
                                      nt * 16 + l16, kc * 32, hl);
                sacc[nt] = wmma32(qf[kc], bf, sacc[nt]);
            }
        }

        // online softmax (per C-row r; row = r + 8*hl)
        float alpha[8];
#pragma unroll
        for (int r = 0; r < 8; r++) {
            float mx = -__builtin_inff();
#pragma unroll
            for (int nt = 0; nt < 4; nt++) mx = fmaxf(mx, sacc[nt][r]);
            mx *= scale;
#pragma unroll
            for (int off = 1; off < 16; off <<= 1)
                mx = fmaxf(mx, __shfl_xor(mx, off, 32));
            float mnew = fmaxf(m[r], mx);
            alpha[r] = __expf(m[r] - mnew);
            float rsum = 0.f;
#pragma unroll
            for (int nt = 0; nt < 4; nt++) {
                float p = __expf(sacc[nt][r] * scale - mnew);
                sacc[nt][r] = p;
                rsum += p;
            }
#pragma unroll
            for (int off = 1; off < 16; off <<= 1)
                rsum += __shfl_xor(rsum, off, 32);
            l[r] = l[r] * alpha[r] + rsum;
            m[r] = mnew;
        }
#pragma unroll
        for (int c = 0; c < 4; c++)
#pragma unroll
            for (int r = 0; r < 8; r++) o[c][r] *= alpha[r];

        // P (f16) -> per-wave LDS row-major so it can be re-read A-layout
#pragma unroll
        for (int nt = 0; nt < 4; nt++)
#pragma unroll
            for (int r = 0; r < 8; r++)
                Ps[wave][r + 8 * hl][nt * 16 + l16] = (f16)sacc[nt][r];

        // PV: 4 dh-tiles x 2 key-chunks
#pragma unroll
        for (int c = 0; c < 4; c++)
#pragma unroll
            for (int kc = 0; kc < 2; kc++) {
                v16h af = load_a_frag(&Ps[wave][0][0], 64, l16, kc * 32, hl);
                v16h bf = load_b_frag(&Vts[cur][0][0], 64,
                                      c * 16 + l16, kc * 32, hl);
                o[c] = wmma32(af, bf, o[c]);
            }
    }

    // normalize + store vals[B*S][E] as f16 (C-layout row = r + 8*hl)
    const int b = bh >> 4, h = bh & (HEADS - 1);
#pragma unroll
    for (int r = 0; r < 8; r++) {
        float inv = 1.0f / l[r];
        int grow = b * SEQ + qblk * 128 + wave * 16 + r + 8 * hl;
#pragma unroll
        for (int c = 0; c < 4; c++) {
            int col = h * DH + c * 16 + l16;
            vals[(size_t)grow * EMBED + col] = (f16)(o[c][r] * inv);
        }
    }
}

// ---------------------------------------------------------------------------
// Output projection: vals[4096,1024] @ Wout (pre-transposed) + b_out -> fp32
// ---------------------------------------------------------------------------
__global__ __launch_bounds__(256) void gemm_out_kernel(
    const f16* __restrict__ X, const f16* __restrict__ Wt,
    const float* __restrict__ bias, float* __restrict__ out,
    int Kdim, int N) {
    __shared__ f16 As[2][TM][TK];
    __shared__ f16 Bs[2][TN][TK];

    const int tid = threadIdx.x;
    const int lane = tid & 31, wave = tid >> 5;
    const int hl = lane >> 4, l16 = lane & 15;
    const int wm = wave >> 2, wn = wave & 3;
    const int bm = blockIdx.x * TM, bn = blockIdx.y * TN;

    v8f acc[4][2];
    const v8f vzero = {0.f, 0.f, 0.f, 0.f, 0.f, 0.f, 0.f, 0.f};
#pragma unroll
    for (int i = 0; i < 4; i++)
#pragma unroll
        for (int j = 0; j < 2; j++) acc[i][j] = vzero;

    stage_tiles(X, Wt, Kdim, bm, bn, 0, tid, As, Bs, 0);

    for (int k0 = 0; k0 < Kdim; k0 += TK) {
        const int cur = (k0 / TK) & 1;
        async_wait0();
        __syncthreads();
        if (k0 + TK < Kdim)
            stage_tiles(X, Wt, Kdim, bm, bn, k0 + TK, tid, As, Bs, cur ^ 1);

        v16h af[4], bf[2];
#pragma unroll
        for (int mt = 0; mt < 4; mt++)
            af[mt] = load_a_frag(&As[cur][0][0], TK,
                                 wm * 64 + mt * 16 + l16, 0, hl);
#pragma unroll
        for (int nt = 0; nt < 2; nt++)
            bf[nt] = load_b_frag(&Bs[cur][0][0], TK,
                                 wn * 32 + nt * 16 + l16, 0, hl);
#pragma unroll
        for (int mt = 0; mt < 4; mt++)
#pragma unroll
            for (int nt = 0; nt < 2; nt++)
                acc[mt][nt] = wmma32(af[mt], bf[nt], acc[mt][nt]);
    }

#pragma unroll
    for (int mt = 0; mt < 4; mt++)
#pragma unroll
        for (int nt = 0; nt < 2; nt++) {
            int col = bn + wn * 32 + nt * 16 + l16;
            float bv = bias[col];
#pragma unroll
            for (int r = 0; r < 8; r++) {
                int row = bm + wm * 64 + mt * 16 + r + 8 * hl;
                out[(size_t)row * N + col] = acc[mt][nt][r] + bv;
            }
        }
}

// ---------------------------------------------------------------------------
extern "C" void kernel_launch(void* const* d_in, const int* in_sizes, int n_in,
                              void* d_out, int out_size, void* d_ws,
                              size_t ws_size, hipStream_t stream) {
    const float* x     = (const float*)d_in[0];   // [2,2048,1024]
    const float* w_qkv = (const float*)d_in[1];   // [1024,3072]
    const float* b_qkv = (const float*)d_in[2];   // [3072]
    const float* w_out = (const float*)d_in[3];   // [1024,1024]
    const float* b_out = (const float*)d_in[4];   // [1024]
    float* out = (float*)d_out;

    char* ws = (char*)d_ws;
    const size_t nx  = (size_t)MROWS * EMBED;         // 4194304
    const size_t nwq = (size_t)EMBED * 3 * EMBED;     // 3145728
    const size_t nwo = (size_t)EMBED * EMBED;         // 1048576
    const size_t nhead = (size_t)BATCH * HEADS * SEQ * DH;  // 4194304

    f16* xh    = (f16*)(ws);
    f16* wqkvt = xh + nx;        // [3072][1024] transposed
    f16* woutt = wqkvt + nwq;    // [1024][1024] transposed
    f16* Qh    = woutt + nwo;    // [B,H,S,DH]
    f16* Kh    = Qh + nhead;     // [B,H,S,DH]
    f16* Vth   = Kh + nhead;     // [B,H,DH,S]  (transposed)
    f16* valsh = Vth + nhead;    // [B*S][E]

    // convert / transpose fp32 -> f16
    f32_to_f16_kernel<<<(int)((nx + 255) / 256), 256, 0, stream>>>(
        x, xh, (int)nx);
    transpose_f32_to_f16_kernel<<<(int)((nwq + 255) / 256), 256, 0, stream>>>(
        w_qkv, wqkvt, EMBED, 3 * EMBED);
    transpose_f32_to_f16_kernel<<<(int)((nwo + 255) / 256), 256, 0, stream>>>(
        w_out, woutt, EMBED, EMBED);

    // QKV projection
    dim3 g1(MROWS / TM, (3 * EMBED) / TN);    // 32 x 24
    gemm_qkv_kernel<<<g1, 256, 0, stream>>>(xh, wqkvt, b_qkv, Qh, Kh, Vth,
                                            EMBED, 3 * EMBED);

    // attention
    dim3 g2(BATCH * HEADS, SEQ / 128);        // 32 x 16
    attn_kernel<<<g2, 256, 0, stream>>>(Qh, Kh, Vth, valsh);

    // output projection
    dim3 g3(MROWS / TM, EMBED / TN);          // 32 x 8
    gemm_out_kernel<<<g3, 256, 0, stream>>>(valsh, woutt, b_out, out,
                                            EMBED, EMBED);
}
